// Attention_23691039605444
// MI455X (gfx1250) — matrix operand order repaired and checked by reading, not compiled, not run on hardware
//
#include <hip/hip_runtime.h>
#include <hip/hip_bf16.h>

typedef __attribute__((ext_vector_type(16))) _Float16 v16h;
typedef __attribute__((ext_vector_type(8)))  _Float16 v8h;
typedef __attribute__((ext_vector_type(8)))  float    v8f;

#define B_   2
#define S_   2048
#define H_   12
#define DM_  768
#define DH_  64
#define IGNORE_ (-100000.0f)

// ---- WMMA helpers -----------------------------------------------------------
// A operand (16xK f16, row-major source): lane holds row (lane%16); its 16
// halves are K = 16*half .. 16*half+15 contiguous -> single 32B load.
static __device__ __forceinline__ v16h load_a16(const _Float16* rowp, int half) {
  return *(const v16h*)(rowp + 16 * half);
}
// B operand (Kx16 f16, column-major source): lane holds column (lane%16),
// K = 16*half .. 16*half+15 contiguous -> single 32B load.
static __device__ __forceinline__ v16h load_b16(const _Float16* colp, int half) {
  return *(const v16h*)(colp + 16 * half);
}
static __device__ __forceinline__ v8f wmma32(v16h a, v16h b, v8f c) {
  return __builtin_amdgcn_wmma_f32_16x16x32_f16(false, a, false, b, (short)0, c,
                                                false, false);
}

// ---- Prep kernels -----------------------------------------------------------
__global__ void cast_f16_k(const float* __restrict__ src,
                           _Float16* __restrict__ dst, int n) {
  int i = blockIdx.x * blockDim.x + threadIdx.x;
  int stride = gridDim.x * blockDim.x;
  for (; i < n; i += stride) dst[i] = (_Float16)src[i];
}

// dst[o][c][r] = (f16) src[o][r][c]
__global__ void transpose_cast_k(const float* __restrict__ src,
                                 _Float16* __restrict__ dst, int outer, int R,
                                 int C) {
  int n = outer * R * C;
  int i = blockIdx.x * blockDim.x + threadIdx.x;
  int stride = gridDim.x * blockDim.x;
  for (; i < n; i += stride) {
    int o = i / (R * C);
    int rem = i - o * R * C;
    int c = rem / R;
    int r = rem - c * R;
    dst[i] = (_Float16)src[(size_t)(o * R + r) * C + c];
  }
}

// ---- QKV projection: one wave computes 16 rows x 64 cols for q,k,v ----------
__global__ __launch_bounds__(32) void qkv_k(
    const _Float16* __restrict__ xh, const _Float16* __restrict__ WqT,
    const _Float16* __restrict__ WkT, const _Float16* __restrict__ WvT,
    const float* __restrict__ bQ, const float* __restrict__ bK,
    const float* __restrict__ bV, _Float16* __restrict__ qh,
    _Float16* __restrict__ kh, _Float16* __restrict__ vT) {
  const int lane = threadIdx.x & 31;
  const int half = lane >> 4;
  const int m16 = lane & 15;
  const int tile = blockIdx.x;  // over B*S/16
  const int h = blockIdx.y;     // head
  const int row0 = tile * 16;
  const int bb = row0 / S_;
  const int srow0 = row0 - bb * S_;

  v8f acc[12];
#pragma unroll
  for (int i = 0; i < 12; ++i) acc[i] = {};

  const _Float16* arow = xh + (size_t)(row0 + m16) * DM_;
  const _Float16* wq = WqT + (size_t)h * DH_ * DM_;
  const _Float16* wk = WkT + (size_t)h * DH_ * DM_;
  const _Float16* wv = WvT + (size_t)h * DH_ * DM_;

  for (int k0 = 0; k0 < DM_; k0 += 32) {
    v16h a = load_a16(arow + k0, half);
#pragma unroll
    for (int t = 0; t < 4; ++t) {
      int col = t * 16 + m16;
      v16h bq = load_b16(wq + (size_t)col * DM_ + k0, half);
      v16h bk = load_b16(wk + (size_t)col * DM_ + k0, half);
      v16h bv = load_b16(wv + (size_t)col * DM_ + k0, half);
      acc[t]     = wmma32(a, bq, acc[t]);
      acc[4 + t] = wmma32(a, bk, acc[4 + t]);
      acc[8 + t] = wmma32(a, bv, acc[8 + t]);
    }
  }
#pragma unroll
  for (int t = 0; t < 4; ++t) {
    int col = t * 16 + m16;
    float biq = bQ[h * DH_ + col];
    float bik = bK[h * DH_ + col];
    float biv = bV[h * DH_ + col];
#pragma unroll
    for (int r = 0; r < 8; ++r) {
      int M = r + 8 * half;
      int sg = srow0 + M;
      size_t qi = (((size_t)bb * H_ + h) * S_ + sg) * DH_ + col;
      qh[qi] = (_Float16)(acc[t][r] + biq);
      kh[qi] = (_Float16)(acc[4 + t][r] + bik);
      size_t vi = (((size_t)bb * H_ + h) * DH_ + col) * S_ + sg;  // V transposed
      vT[vi] = (_Float16)(acc[8 + t][r] + biv);
    }
  }
}

// ---- Flash attention: one wave per 16-query tile, 64-key blocks -------------
__global__ __launch_bounds__(32) void attn_k(const _Float16* __restrict__ qh,
                                             const _Float16* __restrict__ kh,
                                             const _Float16* __restrict__ vT,
                                             _Float16* __restrict__ zh) {
  __shared__ _Float16 P[16 * 64];  // C-layout -> A-layout bounce for probs
  const int lane = threadIdx.x & 31;
  const int half = lane >> 4;
  const int m16 = lane & 15;
  const int q0 = blockIdx.x * 16;
  const int bh = blockIdx.y;
  const int bb = bh / H_;
  const int h = bh - bb * H_;

  const _Float16* qhead = qh + ((size_t)bb * H_ + h) * S_ * DH_;
  const _Float16* khead = kh + ((size_t)bb * H_ + h) * S_ * DH_;
  const _Float16* vhead = vT + ((size_t)bb * H_ + h) * DH_ * S_;

  const _Float16* qrow = qhead + (size_t)(q0 + m16) * DH_;
  v16h aq0 = load_a16(qrow, half);       // K = 0..31 of d_head
  v16h aq1 = load_a16(qrow + 32, half);  // K = 32..63

  v8f o[4];
#pragma unroll
  for (int i = 0; i < 4; ++i) o[i] = {};
  float mrow[8], lrow[8];
#pragma unroll
  for (int r = 0; r < 8; ++r) { mrow[r] = -3.0e38f; lrow[r] = 0.0f; }

  const int nb = q0 / 64 + 1;  // key blocks needed for causal coverage
  for (int blk = 0; blk < nb; ++blk) {
    const int kb = blk * 64;
    v8f st[4];
#pragma unroll
    for (int nt = 0; nt < 4; ++nt) {
      int key = kb + nt * 16 + m16;
      const _Float16* kcol = khead + (size_t)key * DH_;  // row of K == column of K^T
      v16h b0 = load_b16(kcol, half);
      v16h b1 = load_b16(kcol + 32, half);
      v8f s = {};
      s = wmma32(aq0, b0, s);
      s = wmma32(aq1, b1, s);
#pragma unroll
      for (int r = 0; r < 8; ++r) {
        float v = s[r] * 0.125f;  // 1/sqrt(64)
        int qg = q0 + r + 8 * half;
        s[r] = (key > qg) ? IGNORE_ : v;  // causal mask
      }
      st[nt] = s;
    }
    // Online softmax per query row (row lives in one 16-lane half).
#pragma unroll
    for (int r = 0; r < 8; ++r) {
      float bmax = fmaxf(fmaxf(st[0][r], st[1][r]), fmaxf(st[2][r], st[3][r]));
      bmax = fmaxf(bmax, __shfl_xor(bmax, 1));
      bmax = fmaxf(bmax, __shfl_xor(bmax, 2));
      bmax = fmaxf(bmax, __shfl_xor(bmax, 4));
      bmax = fmaxf(bmax, __shfl_xor(bmax, 8));
      float mnew = fmaxf(mrow[r], bmax);
      float corr = __expf(mrow[r] - mnew);
      mrow[r] = mnew;
      float psum = 0.0f;
#pragma unroll
      for (int nt = 0; nt < 4; ++nt) {
        float p = __expf(st[nt][r] - mnew);
        st[nt][r] = p;
        psum += p;
      }
      psum += __shfl_xor(psum, 1);
      psum += __shfl_xor(psum, 2);
      psum += __shfl_xor(psum, 4);
      psum += __shfl_xor(psum, 8);
      lrow[r] = lrow[r] * corr + psum;
#pragma unroll
      for (int dt = 0; dt < 4; ++dt) o[dt][r] *= corr;
    }
    // C-layout probs -> LDS (row-major 16x64) -> reload in A-layout.
    __syncthreads();
#pragma unroll
    for (int nt = 0; nt < 4; ++nt)
#pragma unroll
      for (int r = 0; r < 8; ++r)
        P[(r + 8 * half) * 64 + nt * 16 + m16] = (_Float16)st[nt][r];
    __syncthreads();
#pragma unroll
    for (int kc = 0; kc < 2; ++kc) {
      v16h ap = load_a16(&P[m16 * 64 + kc * 32], half);
#pragma unroll
      for (int dt = 0; dt < 4; ++dt) {
        const _Float16* vcol =
            vhead + (size_t)(dt * 16 + m16) * S_ + kb + kc * 32;
        v16h bv = load_b16(vcol, half);
        o[dt] = wmma32(ap, bv, o[dt]);
      }
    }
  }
  // Normalize and emit z in [B*S][H*DH] f16 layout for the output projection.
#pragma unroll
  for (int dt = 0; dt < 4; ++dt) {
    int col = dt * 16 + m16;
#pragma unroll
    for (int r = 0; r < 8; ++r) {
      int M = r + 8 * half;
      float z = o[dt][r] / lrow[r];
      size_t zi = ((size_t)bb * S_ + q0 + M) * DM_ + (size_t)h * DH_ + col;
      zh[zi] = (_Float16)z;
    }
  }
}

// ---- Output projection: 16 rows x 64 cols per wave, fp32 out ----------------
__global__ __launch_bounds__(32) void proj_k(const _Float16* __restrict__ zh,
                                             const _Float16* __restrict__ WoT,
                                             const float* __restrict__ bO,
                                             float* __restrict__ out) {
  const int lane = threadIdx.x & 31;
  const int half = lane >> 4;
  const int m16 = lane & 15;
  const int row0 = blockIdx.x * 16;
  const int n0 = blockIdx.y * 64;
  v8f acc[4];
#pragma unroll
  for (int i = 0; i < 4; ++i) acc[i] = {};
  const _Float16* arow = zh + (size_t)(row0 + m16) * DM_;
  for (int k0 = 0; k0 < DM_; k0 += 32) {
    v16h a = load_a16(arow + k0, half);
#pragma unroll
    for (int t = 0; t < 4; ++t) {
      v16h b = load_b16(WoT + (size_t)(n0 + t * 16 + m16) * DM_ + k0, half);
      acc[t] = wmma32(a, b, acc[t]);
    }
  }
#pragma unroll
  for (int t = 0; t < 4; ++t) {
    int col = n0 + t * 16 + m16;
    float bias = bO[col];
#pragma unroll
    for (int r = 0; r < 8; ++r)
      out[(size_t)(row0 + r + 8 * half) * DM_ + col] = acc[t][r] + bias;
  }
}

// ---- Host launch ------------------------------------------------------------
extern "C" void kernel_launch(void* const* d_in, const int* in_sizes, int n_in,
                              void* d_out, int out_size, void* d_ws,
                              size_t ws_size, hipStream_t stream) {
  (void)in_sizes; (void)n_in; (void)out_size; (void)ws_size;
  const float* x  = (const float*)d_in[0];
  const float* WQ = (const float*)d_in[1];
  const float* WK = (const float*)d_in[2];
  const float* WV = (const float*)d_in[3];
  const float* bQ = (const float*)d_in[4];
  const float* bK = (const float*)d_in[5];
  const float* bV = (const float*)d_in[6];
  const float* WO = (const float*)d_in[7];
  const float* bO = (const float*)d_in[8];
  float* out = (float*)d_out;

  char* ws = (char*)d_ws;
  size_t off = 0;
  auto alloc = [&](size_t bytes) -> _Float16* {
    _Float16* p = (_Float16*)(ws + off);
    off += (bytes + 255) & ~(size_t)255;
    return p;
  };
  const size_t nx = (size_t)B_ * S_ * DM_;
  const size_t nw = (size_t)H_ * DM_ * DH_;
  const size_t nqkv = (size_t)B_ * H_ * S_ * DH_;
  _Float16* xh  = alloc(nx * 2);
  _Float16* WqT = alloc(nw * 2);
  _Float16* WkT = alloc(nw * 2);
  _Float16* WvT = alloc(nw * 2);
  _Float16* WoT = alloc((size_t)DM_ * DM_ * 2);
  _Float16* qhb = alloc(nqkv * 2);
  _Float16* khb = alloc(nqkv * 2);
  _Float16* vTb = alloc(nqkv * 2);
  _Float16* zhb = alloc(nx * 2);

  cast_f16_k<<<1024, 256, 0, stream>>>(x, xh, (int)nx);
  transpose_cast_k<<<512, 256, 0, stream>>>(WQ, WqT, H_, DM_, DH_);
  transpose_cast_k<<<512, 256, 0, stream>>>(WK, WkT, H_, DM_, DH_);
  transpose_cast_k<<<512, 256, 0, stream>>>(WV, WvT, H_, DM_, DH_);
  transpose_cast_k<<<512, 256, 0, stream>>>(WO, WoT, 1, DM_, DM_);

  qkv_k<<<dim3(B_ * S_ / 16, H_), 32, 0, stream>>>(xh, WqT, WkT, WvT, bQ, bK,
                                                   bV, qhb, khb, vTb);
  attn_k<<<dim3(S_ / 16, B_ * H_), 32, 0, stream>>>(qhb, khb, vTb, zhb);
  proj_k<<<dim3(B_ * S_ / 16, DM_ / 64), 32, 0, stream>>>(zhb, WoT, bO, out);
}
